// FrameAugment_89034672046362
// MI455X (gfx1250) — matrix operand order repaired and checked by reading, not compiled, not run on hardware
//
#include <hip/hip_runtime.h>

// ---------------------------------------------------------------------------
// FrameAugment fused softmax + GEMM for MI455X (gfx1250, wave32, WMMA + TDM).
//
//   out[b,r,f] = sum_s softmax_s((-|flex[r]-fixed[s]| + gumbel(u[b,r,s]))/T)
//                * feature[b,s,f]        B=32, R=1536, S=2048, F=1024, T=0.2
//
// Pipeline:
//   K1: feature f32 -> bf16, transposed per 32-row K-slab into d_ws
//       (one-time, memory-bound).
//   K2: per (b, 16-row tile): exact f32 softmax -> bf16 path in LDS, then
//       64 K-stages of v_wmma_f32_16x16x32_bf16; each stage's 64KB feature
//       slab is DMA'd by the Tensor Data Mover (tensor_load_to_lds) with
//       hardware padding (8B every 64B -> conflict-free stride-36 LDS rows),
//       double-buffered so the DMA overlaps the WMMAs (TENSORcnt + barriers).
// ---------------------------------------------------------------------------

typedef __attribute__((ext_vector_type(16))) __bf16        v16bf;
typedef __attribute__((ext_vector_type(8)))  float         v8f;
typedef __attribute__((ext_vector_type(4)))  unsigned int  u32x4;
typedef __attribute__((ext_vector_type(2)))  unsigned int  u32x2;
typedef __attribute__((ext_vector_type(4)))  float         f32x4;
typedef __attribute__((ext_vector_type(4)))  unsigned int  uv4;
typedef __attribute__((ext_vector_type(8)))  int           iv8;
typedef __attribute__((ext_vector_type(4)))  int           iv4;

struct A2 { u32x4 lo, hi; };            // 32B -> v16bf (A fragment)
struct B4 { u32x2 a, b, c, d; };        // 32B -> v16bf (B fragment)

#define BATCH   32
#define SEQ     2048
#define RED     1536
#define FEAT    1024
#define INV_T   5.0f
#define EPS_F   1e-12f

#define KB      32            // K rows per staged slab
#define NSLAB   (SEQ / KB)    // 64 slabs
#define SLAB_HALVES (FEAT * KB)         // 32768 bf16 = 64KB per slab in ws
#define SMA_STR 2056          // path row stride in halves (2048 + 8 pad)
#define SMB_STR 36            // feature^T row stride in halves (32 + 4 pad)
                              // 72B rows -> bank step 18 -> conflict-free

__device__ __forceinline__ unsigned short f2bf(float f) {
    unsigned int u = __builtin_bit_cast(unsigned int, f);
    u += 0x7FFFu + ((u >> 16) & 1u);
    return (unsigned short)(u >> 16);
}
__device__ __forceinline__ unsigned int pack_bf2(float a, float b) {
    return (unsigned int)f2bf(a) | ((unsigned int)f2bf(b) << 16);
}

// ---------------------------------------------------------------------------
// TDM: DMA one 64KB contiguous bf16 slab from ws into LDS, inserting
// 2 DWORDs of padding every 16 DWORDs (-> 36-half LDS row stride).
// D# fields per CDNA5 ISA (1-D tile of 32768 halves).
// ---------------------------------------------------------------------------
__device__ __forceinline__ void tdm_load_slab(const void* gaddr,
                                              unsigned lds_byte_addr) {
    unsigned long long ga = (unsigned long long)(__SIZE_TYPE__)gaddr;
    uv4 g0 = (uv4)0u;
    g0[0] = 1u;                                     // count=1, user D#
    g0[1] = lds_byte_addr;                          // lds_addr
    g0[2] = (unsigned)ga;                           // global_addr[31:0]
    g0[3] = (unsigned)((ga >> 32) & 0x01FFFFFFu)    // global_addr[56:32]
          | (2u << 30);                             // type = 2 (image)
    iv8 g1 = (iv8)0;
    g1[0] = (int)((1u << 16)      // data_size = 1 (2 bytes)
                | (1u << 20)      // pad_enable
                | (3u << 22)      // pad_interval: 16 DWORDs (64B)
                | (1u << 25));    // pad_amount:   2 DWORDs (8B)
    g1[1] = (int)0x80000000u;     // tensor_dim0 = 32768 (lo16 in [31:16])
    g1[2] = (int)(1u << 16);      // tensor_dim1 = 1
    g1[3] = (int)0x80000000u;     // tile_dim0 = 32768 in [31:16]
    g1[4] = 1;                    // tile_dim1 = 1
    g1[5] = 32768;                // tensor_dim0_stride (lo32)
    iv4 g2 = (iv4)0;
    iv4 g3 = (iv4)0;
#if __has_include(<hip/amd_detail/amd_gfx1250_TDM.h>)
    __builtin_amdgcn_tensor_load_to_lds(g0, g1, g2, g3, (iv8)0, 0); // 6-arg
#else
    __builtin_amdgcn_tensor_load_to_lds(g0, g1, g2, g3, 0);         // 5-arg
#endif
}

// ---------------------------------------------------------------------------
// K1: feature[b, kb..kb+31, :] f32 -> bf16 transposed [n][k] slab in ws.
// Reads coalesced b128; transposes through 64KB LDS; writes ws coalesced.
// ---------------------------------------------------------------------------
__global__ __launch_bounds__(256)
void feat_pack_kernel(const float* __restrict__ feature,
                      unsigned short* __restrict__ ws) {
    __shared__ __align__(16) unsigned short t[FEAT * KB];  // 64KB, [n][k]
    const int tid  = threadIdx.x;
    const int slab = blockIdx.x;
    const int b    = blockIdx.y;
    const int n4   = tid * 4;

    const float* __restrict__ fb =
        feature + ((size_t)b * SEQ + slab * KB) * (size_t)FEAT + n4;
    #pragma unroll
    for (int rb = 0; rb < 8; ++rb) {
        const f32x4 r0 = *(const f32x4*)(fb + (rb * 4 + 0) * FEAT);
        const f32x4 r1 = *(const f32x4*)(fb + (rb * 4 + 1) * FEAT);
        const f32x4 r2 = *(const f32x4*)(fb + (rb * 4 + 2) * FEAT);
        const f32x4 r3 = *(const f32x4*)(fb + (rb * 4 + 3) * FEAT);
        #pragma unroll
        for (int j = 0; j < 4; ++j) {
            u32x2 w;
            w.x = pack_bf2(r0[j], r1[j]);
            w.y = pack_bf2(r2[j], r3[j]);
            *(u32x2*)&t[(n4 + j) * KB + rb * 4] = w;
        }
    }
    __syncthreads();
    // LDS layout == ws layout -> straight coalesced b128 copy out.
    const u32x4* src = (const u32x4*)t;
    u32x4* dst = (u32x4*)(ws + ((size_t)b * NSLAB + slab) * SLAB_HALVES);
    #pragma unroll
    for (int i = 0; i < 16; ++i)
        dst[tid + i * 256] = src[tid + i * 256];
}

// ---------------------------------------------------------------------------
// K2: fused softmax + WMMA GEMM, feature slabs DMA'd by the TDM.
// ---------------------------------------------------------------------------
__global__ __launch_bounds__(256)
void frameaug_tdm_kernel(const float* __restrict__ fixedn,     // [S]
                         const float* __restrict__ flexn,      // [R]
                         const float* __restrict__ gumbel_u,   // [B,R,S]
                         const unsigned short* __restrict__ ws,// packed bf16
                         float* __restrict__ out)              // [B,R,F]
{
    __shared__ __align__(16) unsigned short smA[16 * SMA_STR];       // 65792B
    __shared__ __align__(16) unsigned short smB[2][FEAT * SMB_STR];  // 2x73728B

    const int tid  = threadIdx.x;
    const int wave = tid >> 5;
    const int lane = tid & 31;
    const int l16  = lane & 15;
    const int kgrp = lane >> 4;
    const int b    = blockIdx.y;
    const int R0   = blockIdx.x * 16;

    const unsigned short* wsb = ws + (size_t)b * NSLAB * SLAB_HALVES;

    // Kick off slab 0 DMA before the softmax so it hides under phase 1.
    if (tid == 0)
        tdm_load_slab(wsb, (unsigned)(__SIZE_TYPE__)&smB[0][0]);

    // ---------------- Phase 1: exact f32 softmax, 2 rows per wave ----------
    for (int rr = 0; rr < 2; ++rr) {
        const int r = wave * 2 + rr;
        const float fx = flexn[R0 + r];
        const float* __restrict__ gurow =
            gumbel_u + ((size_t)b * RED + (R0 + r)) * (size_t)SEQ;

        float x[64];
        float m = -3.0e38f;
        #pragma unroll
        for (int i = 0; i < 64; ++i) {
            const int s = i * 32 + lane;
            const float u  = gurow[s];
            const float g  = -__logf(-__logf(u + EPS_F) + EPS_F);
            const float lg = -fabsf(fx - fixedn[s]);
            const float v  = (lg + g) * INV_T;
            x[i] = v;
            m = fmaxf(m, v);
        }
        #pragma unroll
        for (int off = 16; off > 0; off >>= 1)
            m = fmaxf(m, __shfl_xor(m, off, 32));
        float sum = 0.0f;
        #pragma unroll
        for (int i = 0; i < 64; ++i) { x[i] = __expf(x[i] - m); sum += x[i]; }
        #pragma unroll
        for (int off = 16; off > 0; off >>= 1)
            sum += __shfl_xor(sum, off, 32);
        const float inv = 1.0f / sum;
        #pragma unroll
        for (int i = 0; i < 64; ++i)
            smA[r * SMA_STR + i * 32 + lane] = f2bf(x[i] * inv);
    }

    // ---------------- Phase 2: double-buffered TDM + WMMA -------------------
    v8f acc[8];
    #pragma unroll
    for (int t = 0; t < 8; ++t) acc[t] = (v8f)(0.0f);

    for (int i = 0; i < NSLAB; ++i) {
        __syncthreads();  // stage i-1 compute done -> buf[(i+1)&1] reusable
        if (tid == 0) {
            __builtin_amdgcn_s_wait_tensorcnt(0);     // slab i landed in LDS
            if (i + 1 < NSLAB)
                tdm_load_slab(wsb + (size_t)(i + 1) * SLAB_HALVES,
                              (unsigned)(__SIZE_TYPE__)&smB[(i + 1) & 1][0]);
        }
        __syncthreads();  // broadcast: buf[i&1] (and smA on i==0) ready

        const int kb = i * KB;
        // A fragment (16-bit A 16x32): lane half kgrp reads
        // halves K = kb+kgrp*16+[0..15] (matches B's contiguous k order).
        const unsigned short* ap = &smA[l16 * SMA_STR + kb + kgrp * 16];
        const u32x4 a_lo = *(const u32x4*)(ap);
        const u32x4 a_hi = *(const u32x4*)(ap + 8);
        const v16bf afrag = __builtin_bit_cast(v16bf, A2{a_lo, a_hi});

        const unsigned short* bbuf = &smB[i & 1][0];
        #pragma unroll
        for (int t = 0; t < 8; ++t) {
            const int n = wave * 128 + t * 16 + l16;
            const unsigned short* bp = bbuf + (size_t)n * SMB_STR + kgrp * 16;
            const u32x2 b0 = *(const u32x2*)(bp + 0);
            const u32x2 b1 = *(const u32x2*)(bp + 4);
            const u32x2 b2 = *(const u32x2*)(bp + 8);
            const u32x2 b3 = *(const u32x2*)(bp + 12);
            const v16bf bfrag = __builtin_bit_cast(v16bf, B4{b0, b1, b2, b3});
            acc[t] = __builtin_amdgcn_wmma_f32_16x16x32_bf16(
                false, afrag, false, bfrag, (short)0, acc[t], false, false);
        }
    }

    // C/D 16x16 layout: lane = N, lanes 16..31 hold M+8, VGPR j = M.
    const int mbase = kgrp * 8;
    const size_t obase =
        ((size_t)b * RED + R0 + mbase) * (size_t)FEAT + wave * 128 + l16;
    #pragma unroll
    for (int t = 0; t < 8; ++t)
        #pragma unroll
        for (int j = 0; j < 8; ++j)
            out[obase + (size_t)j * FEAT + t * 16] = acc[t][j];
}

// ---------------------------------------------------------------------------
// Fallback (self-contained, no workspace): round-1 kernel with in-block
// feature conversion. Used only when ws_size is too small.
// ---------------------------------------------------------------------------
__global__ __launch_bounds__(256)
void frameaug_fused_kernel(const float* __restrict__ feature,
                           const float* __restrict__ fixedn,
                           const float* __restrict__ flexn,
                           const float* __restrict__ gumbel_u,
                           float* __restrict__ out)
{
    __shared__ __align__(16) unsigned short smA[16 * SMA_STR];
    __shared__ __align__(16) unsigned short smB[FEAT * SMB_STR];

    const int tid  = threadIdx.x;
    const int wave = tid >> 5;
    const int lane = tid & 31;
    const int l16  = lane & 15;
    const int kgrp = lane >> 4;
    const int b    = blockIdx.y;
    const int R0   = blockIdx.x * 16;

    for (int rr = 0; rr < 2; ++rr) {
        const int r = wave * 2 + rr;
        const float fx = flexn[R0 + r];
        const float* __restrict__ gurow =
            gumbel_u + ((size_t)b * RED + (R0 + r)) * (size_t)SEQ;
        float x[64];
        float m = -3.0e38f;
        #pragma unroll
        for (int i = 0; i < 64; ++i) {
            const int s = i * 32 + lane;
            const float u  = gurow[s];
            const float g  = -__logf(-__logf(u + EPS_F) + EPS_F);
            const float v  = (-fabsf(fx - fixedn[s]) + g) * INV_T;
            x[i] = v;
            m = fmaxf(m, v);
        }
        #pragma unroll
        for (int off = 16; off > 0; off >>= 1)
            m = fmaxf(m, __shfl_xor(m, off, 32));
        float sum = 0.0f;
        #pragma unroll
        for (int i = 0; i < 64; ++i) { x[i] = __expf(x[i] - m); sum += x[i]; }
        #pragma unroll
        for (int off = 16; off > 0; off >>= 1)
            sum += __shfl_xor(sum, off, 32);
        const float inv = 1.0f / sum;
        #pragma unroll
        for (int i = 0; i < 64; ++i)
            smA[r * SMA_STR + i * 32 + lane] = f2bf(x[i] * inv);
    }

    v8f acc[8];
    #pragma unroll
    for (int t = 0; t < 8; ++t) acc[t] = (v8f)(0.0f);
    const int n4 = tid * 4;

    for (int kb = 0; kb < SEQ; kb += KB) {
        __syncthreads();
        const float* __restrict__ fb =
            feature + ((size_t)b * SEQ + kb) * (size_t)FEAT + n4;
        #pragma unroll
        for (int rb = 0; rb < 8; ++rb) {
            const f32x4 r0 = *(const f32x4*)(fb + (rb * 4 + 0) * FEAT);
            const f32x4 r1 = *(const f32x4*)(fb + (rb * 4 + 1) * FEAT);
            const f32x4 r2 = *(const f32x4*)(fb + (rb * 4 + 2) * FEAT);
            const f32x4 r3 = *(const f32x4*)(fb + (rb * 4 + 3) * FEAT);
            #pragma unroll
            for (int j = 0; j < 4; ++j) {
                u32x2 w;
                w.x = pack_bf2(r0[j], r1[j]);
                w.y = pack_bf2(r2[j], r3[j]);
                *(u32x2*)&smB[(size_t)(n4 + j) * SMB_STR + rb * 4] = w;
            }
        }
        __syncthreads();

        const unsigned short* ap = &smA[l16 * SMA_STR + kb + kgrp * 16];
        const u32x4 a_lo = *(const u32x4*)(ap);
        const u32x4 a_hi = *(const u32x4*)(ap + 8);
        const v16bf afrag = __builtin_bit_cast(v16bf, A2{a_lo, a_hi});
        #pragma unroll
        for (int t = 0; t < 8; ++t) {
            const int n = wave * 128 + t * 16 + l16;
            const unsigned short* bp = &smB[(size_t)n * SMB_STR + kgrp * 16];
            const u32x2 b0 = *(const u32x2*)(bp + 0);
            const u32x2 b1 = *(const u32x2*)(bp + 4);
            const u32x2 b2 = *(const u32x2*)(bp + 8);
            const u32x2 b3 = *(const u32x2*)(bp + 12);
            const v16bf bfrag = __builtin_bit_cast(v16bf, B4{b0, b1, b2, b3});
            acc[t] = __builtin_amdgcn_wmma_f32_16x16x32_bf16(
                false, afrag, false, bfrag, (short)0, acc[t], false, false);
        }
    }

    const int mbase = kgrp * 8;
    const size_t obase =
        ((size_t)b * RED + R0 + mbase) * (size_t)FEAT + wave * 128 + l16;
    #pragma unroll
    for (int t = 0; t < 8; ++t)
        #pragma unroll
        for (int j = 0; j < 8; ++j)
            out[obase + (size_t)j * FEAT + t * 16] = acc[t][j];
}

extern "C" void kernel_launch(void* const* d_in, const int* in_sizes, int n_in,
                              void* d_out, int out_size, void* d_ws, size_t ws_size,
                              hipStream_t stream) {
    (void)in_sizes; (void)n_in; (void)out_size;
    const float* feature = (const float*)d_in[0];
    const float* fixedn  = (const float*)d_in[1];
    const float* flexn   = (const float*)d_in[2];
    const float* gu      = (const float*)d_in[3];
    float* out           = (float*)d_out;

    dim3 grid(RED / 16, BATCH);   // r-tiles fastest -> feature[b] L2 reuse
    const size_t WS_NEED = (size_t)BATCH * SEQ * FEAT * sizeof(unsigned short);
    if (ws_size >= WS_NEED) {
        unsigned short* ws = (unsigned short*)d_ws;
        feat_pack_kernel<<<dim3(NSLAB, BATCH), 256, 0, stream>>>(feature, ws);
        frameaug_tdm_kernel<<<grid, 256, 0, stream>>>(fixedn, flexn, gu, ws, out);
    } else {
        frameaug_fused_kernel<<<grid, 256, 0, stream>>>(feature, fixedn, flexn,
                                                        gu, out);
    }
}
